// GAT_72619307040791
// MI455X (gfx1250) — matrix operand order repaired and checked by reading, not compiled, not run on hardware
//
#include <hip/hip_runtime.h>
#include <hip/hip_bf16.h>

typedef __attribute__((ext_vector_type(16))) _Float16 v16h;
typedef __attribute__((ext_vector_type(8)))  float    v8f;

#define SLOPE 0.3f
#define BN_EPS 1e-5f

// ---------- monotone float<->uint key for atomic max over floats ----------
__device__ __forceinline__ unsigned fkey(float f) {
    unsigned u = __float_as_uint(f);
    return (u & 0x80000000u) ? ~u : (u | 0x80000000u);
}
__device__ __forceinline__ float fdecode(unsigned k) {
    unsigned u = (k & 0x80000000u) ? (k & 0x7FFFFFFFu) : ~k;
    return __uint_as_float(u);
}

// ---------- weight convert: Wt[n*128+k] = (f16) W[k*128+n] ----------
__global__ void cvt_w_kernel(const float* __restrict__ W, _Float16* __restrict__ Wt) {
    int n = blockIdx.x, k = threadIdx.x;
    Wt[n * 128 + k] = (_Float16)W[(size_t)k * 128 + n];
}

// ---------- WMMA GEMM: C[M,128] = A[M,128] @ W[128,128], Wt is f16 transposed ----------
// grid.x = M/16 (M multiple of 16), block = 256 (8 waves); wave w -> n-tile w.
__global__ void __launch_bounds__(256)
gemm128_wmma(const float* __restrict__ A, const _Float16* __restrict__ Wt,
             float* __restrict__ C) {
    const int mtile = blockIdx.x;
    const int wv    = threadIdx.x >> 5;
    const int lane  = threadIdx.x & 31;
    const int half  = lane >> 4;      // 0 or 1
    const int l15   = lane & 15;

    const int mrow = mtile * 16 + l15;          // A-frag row for this lane
    const int klo  = half * 16;                 // A-frag K sub-offset (contiguous k = 16*half + s)
    const int ncol = wv * 16 + l15;             // B-frag column for this lane

    v8f acc = {};
#pragma unroll
    for (int ks = 0; ks < 4; ++ks) {
        const int k0 = ks * 32;
        // A fragment: contiguous k order (same as B) -> lanes 0-15 K{0..15}, lanes 16-31 K{16..31}
        const float* ap = A + (size_t)mrow * 128 + k0 + klo;
        float4 a0 = *(const float4*)(ap);
        float4 a1 = *(const float4*)(ap + 4);
        float4 a2 = *(const float4*)(ap + 8);
        float4 a3 = *(const float4*)(ap + 12);
        v16h af;
        af[0]  = (_Float16)a0.x; af[1]  = (_Float16)a0.y; af[2]  = (_Float16)a0.z; af[3]  = (_Float16)a0.w;
        af[4]  = (_Float16)a1.x; af[5]  = (_Float16)a1.y; af[6]  = (_Float16)a1.z; af[7]  = (_Float16)a1.w;
        af[8]  = (_Float16)a2.x; af[9]  = (_Float16)a2.y; af[10] = (_Float16)a2.z; af[11] = (_Float16)a2.w;
        af[12] = (_Float16)a3.x; af[13] = (_Float16)a3.y; af[14] = (_Float16)a3.z; af[15] = (_Float16)a3.w;
        // B fragment: lane n holds K = k0 + (half?16:0) .. +15 for column ncol (Wt is [n][k], contiguous)
        const _Float16* bp = Wt + (size_t)ncol * 128 + k0 + half * 16;
        v16h bf = *(const v16h*)bp;
        acc = __builtin_amdgcn_wmma_f32_16x16x32_f16(false, af, false, bf,
                                                     (short)0, acc, false, false);
    }
    // D layout: lane<16 -> N=lane, M=r ; lane>=16 -> N=lane-16, M=8+r
    float* cp = C + ((size_t)(mtile * 16 + half * 8)) * 128 + wv * 16 + l15;
#pragma unroll
    for (int r = 0; r < 8; ++r) cp[(size_t)r * 128] = acc[r];
}

// ---------- edge pass 1: per-edge per-head score + segment max ----------
// one wave32 per edge, 4 channels per lane; head group = C/4 lanes
template <int H, int C>
__global__ void __launch_bounds__(256)
edge_score(const float* __restrict__ xl, const float* __restrict__ xr,
           const float* __restrict__ att, const int* __restrict__ src,
           const int* __restrict__ dst, float* __restrict__ score,
           unsigned* __restrict__ mkey, int E) {
    constexpr int GRP = C / 4;
    int e = blockIdx.x * 8 + (threadIdx.x >> 5);
    if (e >= E) return;
    const int lane = threadIdx.x & 31;
    const int c0 = lane * 4;
    const int s = src[e], d = dst[e];
    float4 a = *(const float4*)(xl + (size_t)s * 128 + c0);
    float4 b = *(const float4*)(xr + (size_t)d * 128 + c0);
    float4 w = *(const float4*)(att + c0);
    float v, acc = 0.f;
    v = a.x + b.x; acc += (v > 0.f ? v : SLOPE * v) * w.x;
    v = a.y + b.y; acc += (v > 0.f ? v : SLOPE * v) * w.y;
    v = a.z + b.z; acc += (v > 0.f ? v : SLOPE * v) * w.z;
    v = a.w + b.w; acc += (v > 0.f ? v : SLOPE * v) * w.w;
#pragma unroll
    for (int off = 1; off < GRP; off <<= 1) acc += __shfl_xor(acc, off);
    const int h = lane / GRP;
    if ((lane & (GRP - 1)) == 0) {
        score[(size_t)e * H + h] = acc;
        atomicMax(&mkey[(size_t)d * H + h], fkey(acc));
    }
}

// ---------- edge pass 2: exp(score - max) and segment sum ----------
template <int H>
__global__ void edge_exp(float* __restrict__ score, const unsigned* __restrict__ mkey,
                         float* __restrict__ ssum, const int* __restrict__ dst, long long EH) {
    long long i = (long long)blockIdx.x * blockDim.x + threadIdx.x;
    if (i >= EH) return;
    long long e = i / H;
    int h = (int)(i - e * H);
    int d = dst[e];
    float ex = __expf(score[i] - fdecode(mkey[(size_t)d * H + h]));
    score[i] = ex;
    unsafeAtomicAdd(&ssum[(size_t)d * H + h], ex);
}

// ---------- edge pass 3: normalize + weighted scatter-add ----------
template <int H, int C>
__global__ void __launch_bounds__(256)
edge_aggr(const float* __restrict__ xl, const float* __restrict__ score,
          const float* __restrict__ ssum, const int* __restrict__ src,
          const int* __restrict__ dst, float* __restrict__ out, int E) {
    constexpr int GRP = C / 4;
    int e = blockIdx.x * 8 + (threadIdx.x >> 5);
    if (e >= E) return;
    const int lane = threadIdx.x & 31;
    const int c0 = lane * 4;
    const int s = src[e], d = dst[e];
    const int h = lane / GRP;
    float w = score[(size_t)e * H + h] / (ssum[(size_t)d * H + h] + 1e-16f);
    float4 a = *(const float4*)(xl + (size_t)s * 128 + c0);
    float* op = out + (size_t)d * 128 + c0;
    unsafeAtomicAdd(op + 0, a.x * w);
    unsafeAtomicAdd(op + 1, a.y * w);
    unsafeAtomicAdd(op + 2, a.z * w);
    unsafeAtomicAdd(op + 3, a.w * w);
}

// ---------- node: +bias, batchnorm(eval), relu (in place) ----------
__global__ void bn_relu(float* __restrict__ h, const float* __restrict__ bias,
                        const float* __restrict__ g, const float* __restrict__ bb,
                        const float* __restrict__ rm, const float* __restrict__ rv,
                        long long total) {
    long long i = (long long)blockIdx.x * blockDim.x + threadIdx.x;
    if (i >= total) return;
    int c = (int)(i & 127);
    float sc = g[c] * rsqrtf(rv[c] + BN_EPS);
    float v = (h[i] + bias[c] - rm[c]) * sc + bb[c];
    h[i] = fmaxf(v, 0.f);
}

// ---------- gate: tanh(gtmp + bg1) . Wg2 per node + per-graph max ----------
__global__ void __launch_bounds__(256)
gate_logit(const float* __restrict__ gtmp, const float* __restrict__ bg1,
           const float* __restrict__ Wg2, const int* __restrict__ batch,
           float* __restrict__ g, unsigned* __restrict__ gm, int N) {
    int n = blockIdx.x * 8 + (threadIdx.x >> 5);
    if (n >= N) return;
    const int lane = threadIdx.x & 31;
    const int c0 = lane * 4;
    float4 t = *(const float4*)(gtmp + (size_t)n * 128 + c0);
    float4 b = *(const float4*)(bg1 + c0);
    float4 w = *(const float4*)(Wg2 + c0);
    float acc = tanhf(t.x + b.x) * w.x + tanhf(t.y + b.y) * w.y +
                tanhf(t.z + b.z) * w.z + tanhf(t.w + b.w) * w.w;
#pragma unroll
    for (int off = 1; off < 32; off <<= 1) acc += __shfl_xor(acc, off);
    if (lane == 0) {
        g[n] = acc;
        atomicMax(&gm[batch[n]], fkey(acc));
    }
}

__global__ void gate_exp(float* __restrict__ g, const unsigned* __restrict__ gm,
                         float* __restrict__ gs, const int* __restrict__ batch, int N) {
    int n = blockIdx.x * blockDim.x + threadIdx.x;
    if (n >= N) return;
    int b = batch[n];
    float ex = __expf(g[n] - fdecode(gm[b]));
    g[n] = ex;
    unsafeAtomicAdd(&gs[b], ex);
}

__global__ void __launch_bounds__(256)
pool_kernel(const float* __restrict__ g, const float* __restrict__ gs,
            const float* __restrict__ h2, const int* __restrict__ batch,
            float* __restrict__ pooled, int N) {
    int n = blockIdx.x * 8 + (threadIdx.x >> 5);
    if (n >= N) return;
    const int lane = threadIdx.x & 31;
    const int c0 = lane * 4;
    int b = batch[n];
    float w = g[n] / (gs[b] + 1e-16f);
    float4 v = *(const float4*)(h2 + (size_t)n * 128 + c0);
    float* op = pooled + (size_t)b * 128 + c0;
    unsafeAtomicAdd(op + 0, v.x * w);
    unsafeAtomicAdd(op + 1, v.y * w);
    unsafeAtomicAdd(op + 2, v.z * w);
    unsafeAtomicAdd(op + 3, v.w * w);
}

// ---------- head: relu(pooled @ Wf1 + bf1) @ Wf2 + bf2 ----------
__global__ void head_fc(const float* __restrict__ pooled, const float* __restrict__ Wf1,
                        const float* __restrict__ bf1, const float* __restrict__ Wf2,
                        const float* __restrict__ bf2, float* __restrict__ out) {
    __shared__ float sh[100];
    int b = blockIdx.x, j = threadIdx.x;
    if (j < 100) {
        float acc = bf1[j];
        for (int k = 0; k < 128; ++k) acc += pooled[(size_t)b * 128 + k] * Wf1[(size_t)k * 100 + j];
        sh[j] = fmaxf(acc, 0.f);
    }
    __syncthreads();
    if (j == 0) {
        float s = bf2[0];
        for (int q = 0; q < 100; ++q) s += sh[q] * Wf2[q];
        out[b] = s;
    }
}

extern "C" void kernel_launch(void* const* d_in, const int* in_sizes, int n_in,
                              void* d_out, int out_size, void* d_ws, size_t ws_size,
                              hipStream_t stream) {
    const int N = in_sizes[0] / 128;   // 50000
    const int E = in_sizes[1] / 2;     // 1600000
    const int G = out_size;            // 500

    const float* x     = (const float*)d_in[0];
    const int*   ei    = (const int*)d_in[1];
    const int*   src   = ei;
    const int*   dst   = ei + E;
    const int*   batch = (const int*)d_in[2];
    const float* Wl1 = (const float*)d_in[3];
    const float* Wr1 = (const float*)d_in[4];
    const float* att1= (const float*)d_in[5];
    const float* b1  = (const float*)d_in[6];
    const float* Wl2 = (const float*)d_in[7];
    const float* Wr2 = (const float*)d_in[8];
    const float* att2= (const float*)d_in[9];
    const float* b2  = (const float*)d_in[10];
    const float* bn_g  = (const float*)d_in[11];
    const float* bn_b  = (const float*)d_in[12];
    const float* bn_rm = (const float*)d_in[13];
    const float* bn_rv = (const float*)d_in[14];
    const float* Wg1 = (const float*)d_in[15];
    const float* bg1 = (const float*)d_in[16];
    const float* Wg2 = (const float*)d_in[17];
    const float* Wf1 = (const float*)d_in[18];
    const float* bf1 = (const float*)d_in[19];
    const float* Wf2 = (const float*)d_in[20];
    const float* bf2 = (const float*)d_in[21];
    float* out = (float*)d_out;

    // workspace carve-out (256B aligned)
    char* w = (char*)d_ws;
    auto carve = [&](size_t bytes) { char* p = w; w += (bytes + 255) & ~(size_t)255; return p; };
    float*    xl     = (float*)carve((size_t)N * 128 * 4);
    float*    xr     = (float*)carve((size_t)N * 128 * 4);
    float*    h1     = (float*)carve((size_t)N * 128 * 4);
    float*    h2     = (float*)carve((size_t)N * 128 * 4);
    float*    alpha  = (float*)carve((size_t)E * 8 * 4);
    unsigned* mk     = (unsigned*)carve((size_t)N * 8 * 4);
    float*    ssum   = (float*)carve((size_t)N * 8 * 4);
    float*    gbuf   = (float*)carve((size_t)N * 4);
    unsigned* gm     = (unsigned*)carve((size_t)G * 4);
    float*    gs     = (float*)carve((size_t)G * 4);
    float*    pooled = (float*)carve((size_t)G * 128 * 4);
    _Float16* w16    = (_Float16*)carve((size_t)5 * 128 * 128 * 2);
    _Float16* Wl1t = w16;
    _Float16* Wr1t = w16 + 16384;
    _Float16* Wl2t = w16 + 2 * 16384;
    _Float16* Wr2t = w16 + 3 * 16384;
    _Float16* Wg1t = w16 + 4 * 16384;

    const int mtiles = N / 16;                 // 50000 % 16 == 0
    const int egrid  = (E + 7) / 8;
    const int ngrid  = (N + 7) / 8;
    const long long NT = (long long)N * 128;

    // weight precision prep
    cvt_w_kernel<<<128, 128, 0, stream>>>(Wl1, Wl1t);
    cvt_w_kernel<<<128, 128, 0, stream>>>(Wr1, Wr1t);
    cvt_w_kernel<<<128, 128, 0, stream>>>(Wl2, Wl2t);
    cvt_w_kernel<<<128, 128, 0, stream>>>(Wr2, Wr2t);
    cvt_w_kernel<<<128, 128, 0, stream>>>(Wg1, Wg1t);

    // ---- layer 1: GATv2, 8 heads x 16 ch ----
    gemm128_wmma<<<mtiles, 256, 0, stream>>>(x, Wl1t, xl);
    gemm128_wmma<<<mtiles, 256, 0, stream>>>(x, Wr1t, xr);
    hipMemsetAsync(mk, 0, (size_t)N * 8 * 4, stream);
    hipMemsetAsync(ssum, 0, (size_t)N * 8 * 4, stream);
    hipMemsetAsync(h1, 0, (size_t)N * 128 * 4, stream);
    edge_score<8, 16><<<egrid, 256, 0, stream>>>(xl, xr, att1, src, dst, alpha, mk, E);
    edge_exp<8><<<(int)(((long long)E * 8 + 255) / 256), 256, 0, stream>>>(alpha, mk, ssum, dst, (long long)E * 8);
    edge_aggr<8, 16><<<egrid, 256, 0, stream>>>(xl, alpha, ssum, src, dst, h1, E);
    bn_relu<<<(int)((NT + 255) / 256), 256, 0, stream>>>(h1, b1, bn_g, bn_b, bn_rm, bn_rv, NT);

    // ---- layer 2: GATv2, 1 head x 128 ch ----
    gemm128_wmma<<<mtiles, 256, 0, stream>>>(h1, Wl2t, xl);
    gemm128_wmma<<<mtiles, 256, 0, stream>>>(h1, Wr2t, xr);
    hipMemsetAsync(mk, 0, (size_t)N * 1 * 4, stream);
    hipMemsetAsync(ssum, 0, (size_t)N * 1 * 4, stream);
    hipMemsetAsync(h2, 0, (size_t)N * 128 * 4, stream);
    edge_score<1, 128><<<egrid, 256, 0, stream>>>(xl, xr, att2, src, dst, alpha, mk, E);
    edge_exp<1><<<(int)(((long long)E + 255) / 256), 256, 0, stream>>>(alpha, mk, ssum, dst, (long long)E);
    edge_aggr<1, 128><<<egrid, 256, 0, stream>>>(xl, alpha, ssum, src, dst, h2, E);
    bn_relu<<<(int)((NT + 255) / 256), 256, 0, stream>>>(h2, b2, bn_g, bn_b, bn_rm, bn_rv, NT);

    // ---- attentional aggregation ----
    gemm128_wmma<<<mtiles, 256, 0, stream>>>(h2, Wg1t, xl);   // reuse xl as gate hidden
    hipMemsetAsync(gm, 0, (size_t)G * 4, stream);
    hipMemsetAsync(gs, 0, (size_t)G * 4, stream);
    hipMemsetAsync(pooled, 0, (size_t)G * 128 * 4, stream);
    gate_logit<<<ngrid, 256, 0, stream>>>(xl, bg1, Wg2, batch, gbuf, gm, N);
    gate_exp<<<(N + 255) / 256, 256, 0, stream>>>(gbuf, gm, gs, batch, N);
    pool_kernel<<<ngrid, 256, 0, stream>>>(gbuf, gs, h2, batch, pooled, N);

    // ---- fc head ----
    head_fc<<<G, 128, 0, stream>>>(pooled, Wf1, bf1, Wf2, bf2, out);

    (void)n_in; (void)ws_size;
}
